// ConditionalDLFactorized17_74680891343537
// MI455X (gfx1250) — matrix operand order repaired and checked by reading, not compiled, not run on hardware
//
#include <hip/hip_runtime.h>
#include <hip/hip_bf16.h>
#include <math.h>

// ---------------- problem constants ----------------
#define T_DIM   2048
#define B_DIM   8
#define C_DIM   512
#define OUT_DIM 512
#define NE      16
#define TOKENS  (T_DIM * B_DIM)      // 16384
#define KTOT    (NE * C_DIM)         // 8192  (fused expert*channel K dimension)
#define LOSS_SCALE 0.01f

typedef __bf16 bf16_t;
typedef __attribute__((ext_vector_type(16))) __bf16 v16bf;
typedef __attribute__((ext_vector_type(8)))  __bf16 v8bf;
typedef __attribute__((ext_vector_type(4)))  __bf16 v4bf;
typedef __attribute__((ext_vector_type(8)))  float  v8f;
typedef __attribute__((ext_vector_type(4)))  unsigned int u32x4;
typedef __attribute__((ext_vector_type(8)))  int i32x8;
typedef __attribute__((ext_vector_type(4)))  int i32x4;

// =====================================================================
// Kernel 1: router — logits = x @ assign_w^T + b, softmax over NE,
// write resp (TOKENS x NE) and accumulate importance (NE).
// One wave per token; assign_w staged in LDS.
// =====================================================================
__global__ __launch_bounds__(256) void router_kernel(
    const float* __restrict__ x, const float* __restrict__ aw,
    const float* __restrict__ ab, float* __restrict__ resp,
    float* __restrict__ imp)
{
    __shared__ float sw[NE * C_DIM];   // 32 KB
    __shared__ float simp[NE];
    const int tid = threadIdx.x;

    #pragma unroll
    for (int p = 0; p < 8; ++p) {
        int idx = (p * 256 + tid) * 4;
        *(float4*)&sw[idx] = *(const float4*)&aw[idx];
    }
    if (tid < NE) simp[tid] = 0.0f;
    __syncthreads();

    const int lane = tid & 31;
    const int wave = tid >> 5;

    float bvals[NE];
    #pragma unroll
    for (int e = 0; e < NE; ++e) bvals[e] = ab[e];

    for (int tt = 0; tt < 8; ++tt) {
        const int token = blockIdx.x * 64 + wave * 8 + tt;

        float4 xv[4];
        #pragma unroll
        for (int s = 0; s < 4; ++s)
            xv[s] = *(const float4*)&x[token * C_DIM + s * 128 + lane * 4];

        float lg[NE];
        #pragma unroll
        for (int e = 0; e < NE; ++e) {
            float p = 0.0f;
            #pragma unroll
            for (int s = 0; s < 4; ++s) {
                float4 wv = *(const float4*)&sw[e * C_DIM + s * 128 + lane * 4];
                p += xv[s].x * wv.x + xv[s].y * wv.y + xv[s].z * wv.z + xv[s].w * wv.w;
            }
            #pragma unroll
            for (int off = 16; off > 0; off >>= 1)
                p += __shfl_xor(p, off, 32);
            lg[e] = p + bvals[e];
        }

        float mx = lg[0];
        #pragma unroll
        for (int e = 1; e < NE; ++e) mx = fmaxf(mx, lg[e]);
        float s = 0.0f;
        #pragma unroll
        for (int e = 0; e < NE; ++e) { lg[e] = __expf(lg[e] - mx); s += lg[e]; }
        const float inv = 1.0f / s;
        #pragma unroll
        for (int e = 0; e < NE; ++e) lg[e] *= inv;

        if (lane == 0) {
            float4 r0 = {lg[0],  lg[1],  lg[2],  lg[3]};
            float4 r1 = {lg[4],  lg[5],  lg[6],  lg[7]};
            float4 r2 = {lg[8],  lg[9],  lg[10], lg[11]};
            float4 r3 = {lg[12], lg[13], lg[14], lg[15]};
            float* rp = resp + (size_t)token * NE;
            *(float4*)(rp + 0)  = r0;
            *(float4*)(rp + 4)  = r1;
            *(float4*)(rp + 8)  = r2;
            *(float4*)(rp + 12) = r3;
            #pragma unroll
            for (int e = 0; e < NE; ++e) atomicAdd(&simp[e], lg[e]);
        }
    }
    __syncthreads();
    if (tid < NE) atomicAdd(&imp[tid], simp[tid]);
}

// =====================================================================
// Kernel 2: loss = 0.01 * std(importance, ddof=1) / mean(importance)
// =====================================================================
__global__ void loss_kernel(const float* __restrict__ imp, float* __restrict__ out_loss)
{
    if (threadIdx.x == 0) {
        float v[NE];
        float s = 0.0f;
        #pragma unroll
        for (int e = 0; e < NE; ++e) { v[e] = imp[e]; s += v[e]; }
        const float mean = s * (1.0f / NE);
        float var = 0.0f;
        #pragma unroll
        for (int e = 0; e < NE; ++e) { float d = v[e] - mean; var += d * d; }
        var *= (1.0f / (NE - 1));
        *out_loss = LOSS_SCALE * sqrtf(var) / mean;
    }
}

// =====================================================================
// Kernel 3: convert pw_w1 (e,o,i) f32 -> Bt[o][e*512+i] bf16.
// =====================================================================
__global__ __launch_bounds__(256) void convert_w_kernel(
    const float* __restrict__ w, bf16_t* __restrict__ bt)
{
    const int t = blockIdx.x * blockDim.x + threadIdx.x;
    const int base = t * 4;
    const int e   = base >> 18;
    const int rem = base & (OUT_DIM * C_DIM - 1);
    const int n   = rem >> 9;
    const int i   = rem & (C_DIM - 1);

    const float4 v = *(const float4*)(w + base);
    v4bf o = { (bf16_t)v.x, (bf16_t)v.y, (bf16_t)v.z, (bf16_t)v.w };
    *(v4bf*)(bt + (size_t)n * KTOT + e * C_DIM + i) = o;
}

// =====================================================================
// Kernel 4: fused MoE GEMM with TDM-fed B tiles.
//   Y(16384 x 512) = A(16384 x 8192) * B,  A[t, e*512+i] = resp[t,e]*x[t,i]
// =====================================================================
#define BM 128
#define BN 128
#define BK 32
#define LDA (BK + 8)                  // 40 halfs (80B row, bank-conflict pad)
#define LDB (BK + 8)                  // TDM pad: every 16 DW insert 4 DW
#define KSTEPS (KTOT / BK)            // 256

// Tensor Data Mover: async copy of one B tile (128 rows x 32 bf16) from
// global (row stride KTOT bf16) into LDS with a 16B pad every 64B row,
// i.e. LDS row stride = LDB halfs. D# layout per CDNA5 ISA ch.8.
__device__ __forceinline__ void tdm_load_b_tile(const bf16_t* gsrc, unsigned lds_addr)
{
    const unsigned long long ga = (unsigned long long)(const void*)gsrc;

    u32x4 g0;
    g0.x = 1u;                                                 // count=1 (valid), user mode
    g0.y = lds_addr;                                           // lds byte address
    g0.z = (unsigned)(ga & 0xffffffffu);                       // global_addr[31:0]
    g0.w = (unsigned)((ga >> 32) & 0x01ffffffu) | (2u << 30);  // global_addr[56:32] | type=2

    i32x8 g1;
    g1[0] = (int)((2u << 16)        // data_size = 4 bytes
                | (1u << 20)        // pad_enable
                | (3u << 22)        // pad_interval: 16 DWORDs (=64B tile row)
                | (3u << 25));      // pad_amount: 4 DWORDs (=16B)
    g1[1] = (int)((4096u & 0xffffu) << 16);                    // tensor_dim0 = 4096 (4B units)
    g1[2] = (int)(((4096u >> 16) & 0xffffu) | ((512u & 0xffffu) << 16)); // tensor_dim1 = 512
    g1[3] = (int)(((512u >> 16) & 0xffffu) | (16u << 16));     // tile_dim0 = 16 (4B units = 64B)
    g1[4] = (int)(128u);                                       // tile_dim1 = 128 rows, tile_dim2 = 0
    g1[5] = (int)(4096u);                                      // tensor_dim0_stride = 4096 (4B units)
    g1[6] = 0;                                                 // stride0[47:32]=0, stride1 lo=0
    g1[7] = 0;                                                 // stride1 hi = 0

    const i32x4 gz4 = {0, 0, 0, 0};                    // 2-D tile: groups 2/3 unused
    const i32x8 gz8 = {0, 0, 0, 0, 0, 0, 0, 0};        // extended group (clang-23 6-arg form)
    __builtin_amdgcn_tensor_load_to_lds(g0, g1, gz4, gz4, gz8, 0);
}

__global__ __launch_bounds__(256) void moe_gemm_kernel(
    const float* __restrict__ x, const float* __restrict__ resp,
    const bf16_t* __restrict__ bt, const float* __restrict__ pwB,
    float* __restrict__ y)
{
    __shared__ bf16_t As[2 * BM * LDA];   // 20.0 KB
    __shared__ bf16_t Bs[2 * BN * LDB];   // 20.0 KB (written by TDM)

    const int tid  = threadIdx.x;
    const int lane = tid & 31;
    const int wave = tid >> 5;
    const int m0 = blockIdx.y * BM;
    const int n0 = blockIdx.x * BN;

    // LDS byte address of Bs (low 32 bits of the generic address = LDS offset).
    // The ptr->int conversion also escapes Bs so its TDM-written reads survive.
    const unsigned bs_base = (unsigned)(size_t)(void*)&Bs[0];
    const bf16_t* bt_blk = bt + (size_t)n0 * KTOT;

    // A-tile global->reg staging map: 8 threads/row * float4, 4 passes
    const int a_col  = (tid & 7) * 4;
    const int a_row0 = tid >> 3;

    float4 a_reg[4];
    float  r_reg[4];

    auto load_regs = [&](int ks) {
        const int k0 = ks * BK;
        const int e  = k0 >> 9;
        const int i0 = k0 & (C_DIM - 1);
        #pragma unroll
        for (int p = 0; p < 4; ++p) {
            const int token = m0 + a_row0 + p * 32;
            a_reg[p] = *(const float4*)&x[token * C_DIM + i0 + a_col];
            r_reg[p] = resp[(size_t)token * NE + e];
        }
    };
    auto store_lds = [&](int buf) {
        bf16_t* a = As + buf * (BM * LDA);
        #pragma unroll
        for (int p = 0; p < 4; ++p) {
            const int row = a_row0 + p * 32;
            const float s = r_reg[p];   // fold resp into A (bf16)
            v4bf v = { (bf16_t)(a_reg[p].x * s), (bf16_t)(a_reg[p].y * s),
                       (bf16_t)(a_reg[p].z * s), (bf16_t)(a_reg[p].w * s) };
            *(v4bf*)&a[row * LDA + a_col] = v;
        }
    };

    const int rw  = (wave >> 2) * 64;
    const int cw  = (wave & 3) * 32;
    const int l16 = lane & 15;
    const int hi  = lane >> 4;

    v8f acc[4][2];
    const v8f vzero = {0.f, 0.f, 0.f, 0.f, 0.f, 0.f, 0.f, 0.f};
    #pragma unroll
    for (int i = 0; i < 4; ++i)
        #pragma unroll
        for (int j = 0; j < 2; ++j) acc[i][j] = vzero;

    // ---- prologue: TDM B tile 0, stage A tile 0 ----
    if (wave == 0) tdm_load_b_tile(bt_blk, bs_base);
    load_regs(0);
    store_lds(0);
    if (wave == 0) __builtin_amdgcn_s_wait_tensorcnt(0);
    __syncthreads();

    for (int ks = 0; ks < KSTEPS; ++ks) {
        const int cur = ks & 1;
        if (ks + 1 < KSTEPS) {
            // nxt buffer is free (consumed last iteration, barrier passed)
            if (wave == 0)
                tdm_load_b_tile(bt_blk + (ks + 1) * BK,
                                bs_base + (unsigned)((cur ^ 1) * (BN * LDB * 2)));
            load_regs(ks + 1);
        }

        const bf16_t* a = As + cur * (BM * LDA);
        const bf16_t* b = Bs + cur * (BN * LDB);
        const int aklo = hi * 8;    // A 16x32: lane<16 K {0..7,16..23}, lane>=16 +8
        const int bklo = hi * 8;    // B 32x16: lane<16 K {0..7,16..23}, lane>=16 +8

        v16bf af[4], bfv[2];
        #pragma unroll
        for (int i = 0; i < 4; ++i) {
            const bf16_t* pa = a + (rw + i * 16 + l16) * LDA + aklo;
            v8bf lo = *(const v8bf*)pa;
            v8bf hh = *(const v8bf*)(pa + 16);
            af[i] = __builtin_shufflevector(lo, hh, 0,1,2,3,4,5,6,7,8,9,10,11,12,13,14,15);
        }
        #pragma unroll
        for (int j = 0; j < 2; ++j) {
            const bf16_t* pb = b + (cw + j * 16 + l16) * LDB + bklo;
            v8bf lo = *(const v8bf*)pb;
            v8bf hh = *(const v8bf*)(pb + 16);
            bfv[j] = __builtin_shufflevector(lo, hh, 0,1,2,3,4,5,6,7,8,9,10,11,12,13,14,15);
        }

        #pragma unroll
        for (int i = 0; i < 4; ++i)
            #pragma unroll
            for (int j = 0; j < 2; ++j)
                acc[i][j] = __builtin_amdgcn_wmma_f32_16x16x32_bf16(
                    false, af[i], false, bfv[j], (short)0, acc[i][j], false, false);

        if (ks + 1 < KSTEPS) store_lds(cur ^ 1);
        if (wave == 0) __builtin_amdgcn_s_wait_tensorcnt(0);  // next B tile resident
        __syncthreads();
    }

    // epilogue: + pw_B, store f32.  C/D layout: VGPR r -> M = hi*8 + r, N = l16
    #pragma unroll
    for (int j = 0; j < 2; ++j) {
        const int col = n0 + cw + j * 16 + l16;
        const float bias = pwB[col];
        #pragma unroll
        for (int i = 0; i < 4; ++i) {
            #pragma unroll
            for (int r = 0; r < 8; ++r) {
                const int row = m0 + rw + i * 16 + hi * 8 + r;
                y[(size_t)row * OUT_DIM + col] = acc[i][j][r] + bias;
            }
        }
    }
}

// =====================================================================
// launch
// =====================================================================
extern "C" void kernel_launch(void* const* d_in, const int* in_sizes, int n_in,
                              void* d_out, int out_size, void* d_ws, size_t ws_size,
                              hipStream_t stream)
{
    (void)in_sizes; (void)n_in; (void)out_size; (void)ws_size;
    const float* x   = (const float*)d_in[0];   // (T,B,C)
    const float* aw  = (const float*)d_in[1];   // (NE,C)
    const float* ab  = (const float*)d_in[2];   // (NE,)
    const float* w1  = (const float*)d_in[3];   // (NE, OUT*C)
    const float* pwB = (const float*)d_in[4];   // (1,1,OUT)

    float* y    = (float*)d_out;
    float* loss = y + (size_t)TOKENS * OUT_DIM;

    char*   ws   = (char*)d_ws;
    float*  resp = (float*)ws;                                       // 1 MB
    float*  imp  = (float*)(ws + (size_t)TOKENS * NE * 4);           // 64 B
    bf16_t* bt   = (bf16_t*)(ws + (size_t)TOKENS * NE * 4 + 4096);   // 8.4 MB

    (void)hipMemsetAsync(imp, 0, NE * sizeof(float), stream);
    router_kernel<<<TOKENS / 64, 256, 0, stream>>>(x, aw, ab, resp, imp);
    loss_kernel<<<1, 32, 0, stream>>>(imp, loss);
    convert_w_kernel<<<(NE * OUT_DIM * C_DIM / 4) / 256, 256, 0, stream>>>(w1, bt);
    moe_gemm_kernel<<<dim3(OUT_DIM / BN, TOKENS / BM), 256, 0, stream>>>(x, resp, bt, pwB, y);
}
